// SequenceLSTMEncoder_32890859553587
// MI455X (gfx1250) — matrix operand order repaired and checked by reading, not compiled, not run on hardware
//
#include <hip/hip_runtime.h>
#include <hip/hip_bf16.h>
#include <math.h>

typedef __attribute__((ext_vector_type(16))) __bf16 v16bf;
typedef __attribute__((ext_vector_type(8)))  float  v8f;

#define T_STEPS  1024
#define BATCH    64
#define FEAT     256
#define HID      512
#define NCOL     2048      // 4*HID gate columns
#define NB_TILES 128       // NCOL / 16
#define NBLOCKS  16
#define NTHREADS 256

struct AB32 { uint4 lo, hi; };   // 32 bytes == v16bf

// ---------------------------------------------------------------------------
// Prep kernel: f32 -> bf16 elementwise (observations)
// ---------------------------------------------------------------------------
__global__ void f32_to_bf16_kernel(const float* __restrict__ src,
                                   __bf16* __restrict__ dst, int n) {
    int gid = blockIdx.x * blockDim.x + threadIdx.x;
    if (gid < n) dst[gid] = (__bf16)src[gid];
}

// ---------------------------------------------------------------------------
// Prep kernel: pack [Wx ; Wh] (K x 2048, f32 row-major) into bf16 tiles laid
// out exactly as the wave32 WMMA B-matrix (32x16, 16-bit) expects:
//   dst[((kb*128 + nb)*32 + lane)*16 + i] = W[kb*32 + (i<8?i:i+8) + (lane>=16?8:0)][nb*16 + lane%16]
// ---------------------------------------------------------------------------
__global__ void pack_weights_kernel(const float* __restrict__ Wx,
                                    const float* __restrict__ Wh,
                                    __bf16* __restrict__ dst,
                                    int F_rows, int KB) {
    int gid = blockIdx.x * blockDim.x + threadIdx.x;
    int total = KB * NB_TILES * 32 * 16;
    if (gid >= total) return;
    int i    = gid & 15;
    int lane = (gid >> 4) & 31;
    int nb   = (gid >> 9) & 127;
    int kb   = gid >> 16;
    int k = kb * 32 + ((i < 8) ? i : i + 8) + ((lane >= 16) ? 8 : 0);
    int n = nb * 16 + (lane & 15);
    float v = (k < F_rows) ? Wx[(size_t)k * NCOL + n]
                           : Wh[(size_t)(k - F_rows) * NCOL + n];
    dst[gid] = (__bf16)v;
}

__global__ void init_barrier_kernel(unsigned* bars) {
    if (threadIdx.x < 2) bars[threadIdx.x] = 0u;
}

// ---------------------------------------------------------------------------
// Sense-reversing grid barrier over NBLOCKS co-resident workgroups.
// ---------------------------------------------------------------------------
__device__ __forceinline__ void grid_barrier(unsigned* count, unsigned* gen) {
    __syncthreads();
    if (threadIdx.x == 0) {
        __threadfence();
        unsigned g = __hip_atomic_load(gen, __ATOMIC_RELAXED, __HIP_MEMORY_SCOPE_AGENT);
        if (atomicAdd(count, 1u) == (unsigned)(NBLOCKS - 1)) {
            __hip_atomic_store(count, 0u, __ATOMIC_RELAXED, __HIP_MEMORY_SCOPE_AGENT);
            __threadfence();
            atomicAdd(gen, 1u);
        } else {
            while (__hip_atomic_load(gen, __ATOMIC_RELAXED, __HIP_MEMORY_SCOPE_AGENT) == g) {
                __builtin_amdgcn_s_sleep(2);
            }
        }
        __threadfence();
    }
    __syncthreads();
}

__device__ __forceinline__ float sigf(float x) {
    return 1.0f / (1.0f + __expf(-x));
}

// A-tile load, 16-bit 16x32 WMMA layout: lane = row M, halves 0-7 <- K=k0..k0+7,
// halves 8-15 <- K=k0+16..k0+23 (k0 += 8 for lanes 16-31): two 16B loads.
__device__ __forceinline__ v16bf load_a16(const __bf16* p) {
    AB32 raw;
    raw.lo = *(const uint4*)(p);
    raw.hi = *(const uint4*)(p + 16);
    return __builtin_bit_cast(v16bf, raw);
}

// Same, but AND every dword with rmask (0 for reset rows, ~0 otherwise):
// branch-free reset of h_{t-1}.
__device__ __forceinline__ v16bf load_a16_masked(const __bf16* p, unsigned rmask) {
    AB32 raw;
    raw.lo = *(const uint4*)(p);
    raw.hi = *(const uint4*)(p + 16);
    raw.lo.x &= rmask; raw.lo.y &= rmask; raw.lo.z &= rmask; raw.lo.w &= rmask;
    raw.hi.x &= rmask; raw.hi.y &= rmask; raw.hi.z &= rmask; raw.hi.w &= rmask;
    return __builtin_bit_cast(v16bf, raw);
}

__device__ __forceinline__ void stage_weights(const __bf16* __restrict__ wpack,
                                              int KB, __bf16* lds_w) {
    // copy this block's 128-column slice; tile = 512 halves = 64 uint4
    int total_u4 = KB * 8 * 64;
    for (int c = threadIdx.x; c < total_u4; c += NTHREADS) {
        int tile = c >> 6, off = c & 63;
        int kb = tile >> 3, nbl = tile & 7;
        const uint4* s = (const uint4*)(wpack +
            (size_t)(kb * NB_TILES + blockIdx.x * 8 + nbl) * 512);
        ((uint4*)(lds_w + (size_t)tile * 512))[off] = s[off];
    }
}

// ---------------------------------------------------------------------------
// One layer of the persistent scan. FK = x-region K-blocks (8 for layer0,
// 16 for layer1); h-region is always HID/32 = 16 K-blocks. LAST selects the
// output sink (bf16 intermediate vs f32 d_out).
// ---------------------------------------------------------------------------
template <int FK, bool LAST>
__device__ __forceinline__ void scan_layer(
    const __bf16* __restrict__ xseq,
    const unsigned char* __restrict__ resets,
    const float* __restrict__ bias,
    __bf16* __restrict__ yout_bf,
    float*  __restrict__ yout_f32,
    float*  __restrict__ zbuf,
    float*  __restrict__ cbuf,
    __bf16* __restrict__ hbuf,
    unsigned* bar_count, unsigned* bar_gen,
    const __bf16* lds_w)
{
    constexpr int HK = HID / 32;          // 16
    constexpr int XSTRIDE = FK * 32;

    const int tid    = threadIdx.x;
    const int wg     = blockIdx.x;
    const int lane   = tid & 31;
    const int wave   = tid >> 5;
    const int mb     = wave & 3;                   // 16-row block
    const int colgrp = wave >> 2;                  // which 4 nb tiles
    const int row    = mb * 16 + (lane & 15);      // A row (M = lane%16)
    const int khalf  = (lane >= 16) ? 8 : 0;       // A-layout K offset, hi lanes
    const int rowhi  = (lane >= 16) ? 8 : 0;       // C/D-layout row offset

    // loop-invariant: per-wave LDS base and bias values
    const __bf16* ldsb = lds_w + (size_t)(colgrp * 4) * 512 + (size_t)lane * 16;
    float bv[4];
    int   col[4];
    #pragma unroll
    for (int j = 0; j < 4; ++j) {
        col[j] = wg * 128 + (colgrp * 4 + j) * 16 + (lane & 15);
        bv[j]  = bias[col[j]];
    }
    const __bf16* hrow = hbuf + (size_t)row * HID;

    for (int t = 0; t < T_STEPS; ++t) {
        const __bf16* xrow = xseq + (size_t)(t * BATCH + row) * XSTRIDE;
        const unsigned rmask =
            ((t == 0) | (resets[t * BATCH + row] != 0)) ? 0u : ~0u;

        // prefetch next step's activation row (global_prefetch_b8)
        if (t + 1 < T_STEPS)
            __builtin_prefetch(xrow + (size_t)BATCH * XSTRIDE, 0, 0);

        // ---------------- phase 1: z = [x;h] @ [Wx;Wh] -----------------------
        v8f acc[4] = {};

        #pragma unroll 4
        for (int kb = 0; kb < FK; ++kb) {                 // x region
            v16bf a = load_a16(xrow + kb * 32 + khalf);
            #pragma unroll
            for (int j = 0; j < 4; ++j) {
                v16bf bm = *(const v16bf*)(ldsb + (size_t)(kb * 8 + j) * 512);
                acc[j] = __builtin_amdgcn_wmma_f32_16x16x32_bf16(
                    false, a, false, bm, (short)0, acc[j], false, false);
            }
        }
        #pragma unroll 4
        for (int kb = 0; kb < HK; ++kb) {                 // h region (masked)
            v16bf a = load_a16_masked(hrow + kb * 32 + khalf, rmask);
            #pragma unroll
            for (int j = 0; j < 4; ++j) {
                v16bf bm = *(const v16bf*)(ldsb + (size_t)((FK + kb) * 8 + j) * 512);
                acc[j] = __builtin_amdgcn_wmma_f32_16x16x32_bf16(
                    false, a, false, bm, (short)0, acc[j], false, false);
            }
        }

        // bias add + scatter z tiles (C/D: lane = col N, VGPR r = row M)
        #pragma unroll
        for (int j = 0; j < 4; ++j) {
            #pragma unroll
            for (int r = 0; r < 8; ++r) {
                int orow = mb * 16 + r + rowhi;
                zbuf[(size_t)orow * NCOL + col[j]] = acc[j][r] + bv[j];
            }
        }

        grid_barrier(bar_count, bar_gen);

        // ---------------- phase 2: gate update -------------------------------
        for (int e = wg * NTHREADS + tid; e < BATCH * HID; e += NBLOCKS * NTHREADS) {
            int b  = e >> 9;             // / HID
            int hh = e & (HID - 1);
            const float zi = zbuf[(size_t)b * NCOL + hh];
            const float zf = zbuf[(size_t)b * NCOL + 512  + hh];
            const float zg = zbuf[(size_t)b * NCOL + 1024 + hh];
            const float zo = zbuf[(size_t)b * NCOL + 1536 + hh];
            bool r2 = (t == 0) || (resets[t * BATCH + b] != 0);
            float c_old = r2 ? 0.0f : cbuf[e];
            float cn = sigf(zf) * c_old + sigf(zi) * tanhf(zg);
            float hn = sigf(zo) * tanhf(cn);
            cbuf[e] = cn;
            hbuf[e] = (__bf16)hn;
            if (LAST)
                yout_f32[(size_t)(t * BATCH + b) * HID + hh] = hn;
            else
                yout_bf[(size_t)(t * BATCH + b) * HID + hh] = (__bf16)hn;
        }

        grid_barrier(bar_count, bar_gen);
    }
}

// ---------------------------------------------------------------------------
// Persistent LSTM scan: both layers, all T steps, one launch.
// grid = 16 blocks x 256 threads (8 wave32). Block owns 128 gate columns,
// weights LDS-resident (192KB / 256KB), wave owns 4 f32 accumulator tiles.
// ---------------------------------------------------------------------------
__launch_bounds__(NTHREADS, 1)
__global__ void lstm_persistent_kernel(
    const __bf16* __restrict__ obs_bf,
    const unsigned char* __restrict__ resets,
    const __bf16* __restrict__ wpack0, const float* __restrict__ bias0,
    const __bf16* __restrict__ wpack1, const float* __restrict__ bias1,
    __bf16* __restrict__ y0,
    float*  __restrict__ out,
    float*  __restrict__ zbuf, float* __restrict__ cbuf,
    __bf16* __restrict__ hbuf,
    unsigned* bar_count, unsigned* bar_gen)
{
    extern __shared__ __bf16 lds_w[];   // [kb][nb_local(8)][lane(32)][i(16)]

    stage_weights(wpack0, (FEAT + HID) / 32, lds_w);
    __syncthreads();
    scan_layer<FEAT / 32, false>(obs_bf, resets, bias0, y0, nullptr,
                                 zbuf, cbuf, hbuf, bar_count, bar_gen, lds_w);

    __syncthreads();
    stage_weights(wpack1, (HID + HID) / 32, lds_w);
    __syncthreads();
    scan_layer<HID / 32, true>(y0, resets, bias1, nullptr, out,
                               zbuf, cbuf, hbuf, bar_count, bar_gen, lds_w);
}

// ---------------------------------------------------------------------------
// Launch
// ---------------------------------------------------------------------------
extern "C" void kernel_launch(void* const* d_in, const int* in_sizes, int n_in,
                              void* d_out, int out_size, void* d_ws, size_t ws_size,
                              hipStream_t stream) {
    (void)in_sizes; (void)n_in; (void)out_size; (void)ws_size;

    const float*         obs    = (const float*)d_in[0];
    const unsigned char* resets = (const unsigned char*)d_in[1];
    const float* Wx0 = (const float*)d_in[2];
    const float* Wh0 = (const float*)d_in[3];
    const float* b0  = (const float*)d_in[4];
    const float* Wx1 = (const float*)d_in[5];
    const float* Wh1 = (const float*)d_in[6];
    const float* b1  = (const float*)d_in[7];
    float* out = (float*)d_out;

    char* ws = (char*)d_ws;
    size_t off = 0;
    auto carve = [&](size_t bytes) -> void* {
        void* p = ws + off;
        off = (off + bytes + 255) & ~(size_t)255;
        return p;
    };
    __bf16* wp0    = (__bf16*)carve((size_t)(FEAT + HID) * NCOL * 2);      // 3 MB
    __bf16* wp1    = (__bf16*)carve((size_t)(HID + HID) * NCOL * 2);       // 4 MB
    __bf16* obs_bf = (__bf16*)carve((size_t)T_STEPS * BATCH * FEAT * 2);   // 32 MB
    __bf16* y0     = (__bf16*)carve((size_t)T_STEPS * BATCH * HID * 2);    // 64 MB
    float*  zbuf   = (float*) carve((size_t)BATCH * NCOL * 4);             // 512 KB
    float*  cbuf   = (float*) carve((size_t)BATCH * HID * 4);              // 128 KB
    __bf16* hbuf   = (__bf16*)carve((size_t)BATCH * HID * 2);              // 64 KB
    unsigned* bars = (unsigned*)carve(256);

    init_barrier_kernel<<<1, 32, 0, stream>>>(bars);

    {
        int n = T_STEPS * BATCH * FEAT;
        f32_to_bf16_kernel<<<(n + 255) / 256, 256, 0, stream>>>(obs, obs_bf, n);
    }
    {
        int kb0 = (FEAT + HID) / 32;  // 24
        int n0 = kb0 * NB_TILES * 32 * 16;
        pack_weights_kernel<<<(n0 + 255) / 256, 256, 0, stream>>>(Wx0, Wh0, wp0, FEAT, kb0);
        int kb1 = (HID + HID) / 32;   // 32
        int n1 = kb1 * NB_TILES * 32 * 16;
        pack_weights_kernel<<<(n1 + 255) / 256, 256, 0, stream>>>(Wx1, Wh1, wp1, HID, kb1);
    }

    const int smem = 32 * 8 * 512 * 2;  // 256 KB: layer-1 weight slice
    (void)hipFuncSetAttribute((const void*)lstm_persistent_kernel,
                              hipFuncAttributeMaxDynamicSharedMemorySize, smem);
    lstm_persistent_kernel<<<NBLOCKS, NTHREADS, smem, stream>>>(
        obs_bf, resets, wp0, b0, wp1, b1, y0, out, zbuf, cbuf, hbuf,
        &bars[0], &bars[1]);
}
